// PBPDenseDensify_55783035240849
// MI455X (gfx1250) — matrix operand order repaired and checked by reading, not compiled, not run on hardware
//
#include <hip/hip_runtime.h>

typedef __attribute__((ext_vector_type(16))) __bf16 v16bf;
typedef __attribute__((ext_vector_type(8)))  float  v8f;
typedef int v4i_t __attribute__((vector_size(16)));   // matches builtin's pointee type

constexpr int DIM = 2048;   // D
constexpr int NB  = 4096;   // BATCH
constexpr int TM = 128, TN = 256, TK = 32;
constexpr int LDT = 40;     // padded K-stride (ushorts); 80B rows keep 16B alignment

struct alignas(16) U128 { unsigned int w[4]; };
union FragU { U128 q[2]; v16bf v; };

#if __has_builtin(__builtin_amdgcn_global_load_async_to_lds_b128)
#define HAVE_ASYNC_LDS 1
#else
#define HAVE_ASYNC_LDS 0
#endif

__device__ __forceinline__ void cp16(unsigned short* l, const unsigned short* g) {
#if HAVE_ASYNC_LDS
  __builtin_amdgcn_global_load_async_to_lds_b128(
      (__attribute__((address_space(1))) v4i_t*)g,
      (__attribute__((address_space(3))) v4i_t*)l, 0, 0);
#else
  *(U128*)l = *(const U128*)g;
#endif
}

__device__ __forceinline__ void wait_stage() {
#if HAVE_ASYNC_LDS
#if __has_builtin(__builtin_amdgcn_s_wait_asynccnt)
  __builtin_amdgcn_s_wait_asynccnt(0);
#else
  asm volatile("s_wait_asynccnt 0" ::: "memory");
#endif
#endif
}

__device__ __forceinline__ unsigned short f2bf(float f) {
  unsigned u = __float_as_uint(f);
  unsigned r = (u >> 16) & 1u;
  u += 0x7fffu + r;                 // round-to-nearest-even
  return (unsigned short)(u >> 16);
}

// ---------------- f32 -> bf16 bulk convert ----------------
__global__ __launch_bounds__(256) void k_tobf16(const float* __restrict__ src,
                                                unsigned short* __restrict__ dst, int n) {
  int i = blockIdx.x * blockDim.x + threadIdx.x;
  int stride = gridDim.x * blockDim.x;
  for (; i < n; i += stride) dst[i] = f2bf(src[i]);
}

// ---------------- row softmax stats: one wave per row ----------------
__global__ __launch_bounds__(256) void k_rowstats(const float* __restrict__ P,
                                                  float* __restrict__ rmax,
                                                  float* __restrict__ rsum) {
  int lane = threadIdx.x & 31, wave = threadIdx.x >> 5;
  int row = blockIdx.x * 8 + wave;
  const float* p = P + (size_t)row * DIM;
  float m = -3.4e38f, s = 0.f;
  for (int i = lane; i < DIM; i += 32) {
    float v = p[i];
    float nm = fmaxf(m, v);
    s = s * __expf(m - nm) + __expf(v - nm);
    m = nm;
  }
  #pragma unroll
  for (int off = 16; off; off >>= 1) {
    float om = __shfl_xor(m, off, 32);
    float os = __shfl_xor(s, off, 32);
    float nm = fmaxf(m, om);
    s = s * __expf(m - nm) + os * __expf(om - nm);
    m = nm;
  }
  if (lane == 0) { rmax[row] = m; rsum[row] = s; }
}

// ---------------- column stats: row-chunked partials + combine ----------------
__global__ __launch_bounds__(256) void k_colpart(const float* __restrict__ P,
                                                 float* __restrict__ pm,
                                                 float* __restrict__ ps) {
  int col = blockIdx.x * 256 + threadIdx.x;
  int rc  = blockIdx.y;
  float m = -3.4e38f, s = 0.f;
  int rbeg = rc * (DIM / 8);
  for (int r = rbeg; r < rbeg + DIM / 8; ++r) {
    float v = P[(size_t)r * DIM + col];
    float nm = fmaxf(m, v);
    s = s * __expf(m - nm) + __expf(v - nm);
    m = nm;
  }
  pm[rc * DIM + col] = m;
  ps[rc * DIM + col] = s;
}

__global__ __launch_bounds__(256) void k_colcombine(const float* __restrict__ pm,
                                                    const float* __restrict__ ps,
                                                    float* __restrict__ cmax,
                                                    float* __restrict__ csum) {
  int col = blockIdx.x * 256 + threadIdx.x;
  float m = -3.4e38f, s = 0.f;
  #pragma unroll
  for (int rc = 0; rc < 8; ++rc) {
    float om = pm[rc * DIM + col], os = ps[rc * DIM + col];
    float nm = fmaxf(m, om);
    s = s * __expf(m - nm) + os * __expf(om - nm);
    m = nm;
  }
  cmax[col] = m; csum[col] = s;
}

// ---------------- perm(P) -> W^T (bf16), tiled transpose ----------------
__global__ __launch_bounds__(256) void k_permw(const float* __restrict__ P,
                                               const float* __restrict__ rmax,
                                               const float* __restrict__ rsum,
                                               const float* __restrict__ cmax,
                                               const float* __restrict__ csum,
                                               unsigned short* __restrict__ Wt) {
  __shared__ unsigned short T[32][33];
  int i0 = blockIdx.x * 32, j0 = blockIdx.y * 32;
  int tid = threadIdx.x;
  #pragma unroll
  for (int e = 0; e < 4; ++e) {
    int idx = tid + e * 256;
    int r = idx >> 5, c = idx & 31;
    float p = P[(size_t)(i0 + r) * DIM + (j0 + c)];
    float w = __expf(2.f * p - rmax[i0 + r] - cmax[j0 + c]) / (rsum[i0 + r] * csum[j0 + c]);
    T[c][r] = f2bf(w);   // transpose in LDS
  }
  __syncthreads();
  #pragma unroll
  for (int e = 0; e < 4; ++e) {
    int idx = tid + e * 256;
    int r = idx >> 5, c = idx & 31;
    Wt[(size_t)(j0 + r) * DIM + (i0 + c)] = T[r][c];  // Wt[n][k] = perm[k][n]
  }
}

// ---------------- compact block-diagonal of B*M: Bc[k*4+c] = B[k][(k&~3)+c] ----------------
__global__ __launch_bounds__(256) void k_bc(const float* __restrict__ B, float* __restrict__ Bc) {
  int i = blockIdx.x * blockDim.x + threadIdx.x;
  if (i >= DIM * 4) return;
  int k = i >> 2, c = i & 3;
  Bc[i] = B[(size_t)k * DIM + (k & ~3) + c];
}

// ---------------- bf16 WMMA GEMM: out = A @ W (W supplied transposed) ----------------
// Block tile 128x256, 8 waves, each wave 64x64 = 4x4 WMMA fragments.
// Double-buffered LDS staging via async global->LDS (ASYNCcnt) when available.
// FINAL=false: epilogue applies block-diagonal Bc, writes bf16 activations.
// FINAL=true : epilogue adds bias + ReLU, writes f32 to d_out.
template <bool FINAL>
__global__ __launch_bounds__(256) void k_gemm(const unsigned short* __restrict__ A,
                                              const unsigned short* __restrict__ Wt,
                                              const float* __restrict__ coef,
                                              void* __restrict__ outv) {
  __shared__ alignas(16) unsigned short Al[2 * TM * LDT];
  __shared__ alignas(16) unsigned short Bl[2 * TN * LDT];
  const int m0 = blockIdx.x * TM, n0 = blockIdx.y * TN;
  const int tid = threadIdx.x, lane = tid & 31, wave = tid >> 5;
  const int wm = (wave & 1) * 64;      // wave M offset in tile
  const int wn = (wave >> 1) * 64;     // wave N offset in tile
  const int half = lane >> 4, l15 = lane & 15;
  const int NK = DIM / TK;             // 64 K-steps

  v8f acc[4][4] = {};

  // stage one K-tile into LDS buffer `buf` (async DMA when available)
  auto stage = [&](int kt, int buf) {
    const int k0 = kt * TK;
    unsigned short* al = Al + buf * (TM * LDT);
    unsigned short* bl = Bl + buf * (TN * LDT);
    #pragma unroll
    for (int i = 0; i < 2; ++i) {      // A tile: 128x32 bf16 = 512 x b128
      int q = tid + i * 256;
      int row = q >> 2, kc = (q & 3) * 8;
      cp16(&al[row * LDT + kc], &A[(size_t)(m0 + row) * DIM + k0 + kc]);
    }
    #pragma unroll
    for (int i = 0; i < 4; ++i) {      // W^T tile: 256x32 bf16 = 1024 x b128
      int q = tid + i * 256;
      int row = q >> 2, kc = (q & 3) * 8;
      cp16(&bl[row * LDT + kc], &Wt[(size_t)(n0 + row) * DIM + k0 + kc]);
    }
  };

  stage(0, 0);
  wait_stage();
  __syncthreads();

  for (int kt = 0; kt < NK; ++kt) {
    const int cur = kt & 1;
    if (kt + 1 < NK) stage(kt + 1, cur ^ 1);   // overlap next-tile DMA with compute

    const unsigned short* al = Al + cur * (TM * LDT);
    const unsigned short* bl = Bl + cur * (TN * LDT);

    // Fragments per ISA §7.12.2 (wave32):
    // A 16x32 bf16: lane m=l&15; half 0 holds K 0..7 & 16..23, half 1 holds K 8..15 & 24..31.
    FragU af[4], bfr[4];
    #pragma unroll
    for (int mf = 0; mf < 4; ++mf) {
      int row = wm + mf * 16 + l15;
      af[mf].q[0] = *(const U128*)&al[row * LDT + 8 * half];
      af[mf].q[1] = *(const U128*)&al[row * LDT + 16 + 8 * half];
    }
    // B 32x16 bf16: lane n=l&15; half 0 holds K 0..7 & 16..23, half 1 holds K 8..15 & 24..31.
    #pragma unroll
    for (int nf = 0; nf < 4; ++nf) {
      int row = wn + nf * 16 + l15;
      bfr[nf].q[0] = *(const U128*)&bl[row * LDT + 8 * half];
      bfr[nf].q[1] = *(const U128*)&bl[row * LDT + 16 + 8 * half];
    }
    #pragma unroll
    for (int mf = 0; mf < 4; ++mf)
      #pragma unroll
      for (int nf = 0; nf < 4; ++nf)
        acc[mf][nf] = __builtin_amdgcn_wmma_f32_16x16x32_bf16(
            false, af[mf].v, false, bfr[nf].v, (short)0, acc[mf][nf], false, false);

    wait_stage();      // next buffer's async DMA has landed
    __syncthreads();   // all waves done reading `cur`; safe to overwrite next iter
  }

  // Epilogue. C/D layout: lane n = l&15, VGPR v holds row m = v + 8*half.
  if (FINAL) {
    float* out = (float*)outv;
    #pragma unroll
    for (int nf = 0; nf < 4; ++nf) {
      int j = n0 + wn + nf * 16 + l15;
      float b = coef[j];
      #pragma unroll
      for (int mf = 0; mf < 4; ++mf) {
        #pragma unroll
        for (int v = 0; v < 8; ++v) {
          int mrow = m0 + wm + mf * 16 + half * 8 + v;
          float s = acc[mf][nf][v] + b;
          out[(size_t)mrow * DIM + j] = s > 0.f ? s : 0.f;
        }
      }
    }
  } else {
    // block-diag: s[m][j] = sum_{r=0..3} out[m][(j&~3)+r] * Bc[(j&~3)+r][j&3]
    // column (j&~3)+r lives in lane (lane&~3)+r (bit4/half preserved by &~3).
    unsigned short* out = (unsigned short*)outv;
    int srcbase = lane & ~3;
    #pragma unroll
    for (int nf = 0; nf < 4; ++nf) {
      int j = n0 + wn + nf * 16 + l15;
      int jb = j & ~3, jc = j & 3;
      float bc0 = coef[(jb + 0) * 4 + jc];
      float bc1 = coef[(jb + 1) * 4 + jc];
      float bc2 = coef[(jb + 2) * 4 + jc];
      float bc3 = coef[(jb + 3) * 4 + jc];
      #pragma unroll
      for (int mf = 0; mf < 4; ++mf) {
        #pragma unroll
        for (int v = 0; v < 8; ++v) {
          float a = acc[mf][nf][v];
          float s = __shfl(a, srcbase + 0, 32) * bc0
                  + __shfl(a, srcbase + 1, 32) * bc1
                  + __shfl(a, srcbase + 2, 32) * bc2
                  + __shfl(a, srcbase + 3, 32) * bc3;
          int mrow = m0 + wm + mf * 16 + half * 8 + v;
          out[(size_t)mrow * DIM + j] = f2bf(s);
        }
      }
    }
  }
}

extern "C" void kernel_launch(void* const* d_in, const int* in_sizes, int n_in,
                              void* d_out, int out_size, void* d_ws, size_t ws_size,
                              hipStream_t stream) {
  (void)in_sizes; (void)n_in; (void)out_size; (void)ws_size;
  const float* x = (const float*)d_in[0];
  const float* P[4] = {(const float*)d_in[1], (const float*)d_in[2],
                       (const float*)d_in[3], (const float*)d_in[4]};
  const float* B[3] = {(const float*)d_in[5], (const float*)d_in[6], (const float*)d_in[7]};
  const float* bias = (const float*)d_in[11];
  float* out = (float*)d_out;

  // workspace carve-up (256B aligned)
  char* ws = (char*)d_ws;
  size_t off = 0;
  auto alloc = [&](size_t bytes) -> void* {
    void* p = ws + off;
    off += (bytes + 255) & ~(size_t)255;
    return p;
  };
  unsigned short* xb = (unsigned short*)alloc((size_t)NB * DIM * 2);   // x in bf16; reused as ping
  unsigned short* W[4];
  for (int i = 0; i < 4; ++i) W[i] = (unsigned short*)alloc((size_t)DIM * DIM * 2);
  unsigned short* act0 = (unsigned short*)alloc((size_t)NB * DIM * 2); // pong
  unsigned short* act1 = xb;                                           // alias: xb dead after GEMM1
  float* Bc[3];
  for (int i = 0; i < 3; ++i) Bc[i] = (float*)alloc((size_t)DIM * 4 * 4);
  float* rmax = (float*)alloc(DIM * 4);
  float* rsum = (float*)alloc(DIM * 4);
  float* cmax = (float*)alloc(DIM * 4);
  float* csum = (float*)alloc(DIM * 4);
  float* pm   = (float*)alloc(8 * DIM * 4);
  float* ps   = (float*)alloc(8 * DIM * 4);

  // 1) x -> bf16
  k_tobf16<<<4096, 256, 0, stream>>>(x, xb, NB * DIM);

  // 2) perm(P_i) -> W_i^T (bf16). Stats buffers reused; stream serializes.
  for (int m = 0; m < 4; ++m) {
    k_rowstats<<<DIM / 8, 256, 0, stream>>>(P[m], rmax, rsum);
    k_colpart<<<dim3(DIM / 256, 8), 256, 0, stream>>>(P[m], pm, ps);
    k_colcombine<<<DIM / 256, 256, 0, stream>>>(pm, ps, cmax, csum);
    k_permw<<<dim3(DIM / 32, DIM / 32), 256, 0, stream>>>(P[m], rmax, rsum, cmax, csum, W[m]);
  }

  // 3) compact block-diagonals
  for (int m = 0; m < 3; ++m)
    k_bc<<<(DIM * 4 + 255) / 256, 256, 0, stream>>>(B[m], Bc[m]);

  // 4) GEMM chain (bf16 WMMA, f32 accumulate)
  dim3 gg(NB / TM, DIM / TN);
  k_gemm<false><<<gg, 256, 0, stream>>>(xb,   W[0], Bc[0], act0);
  k_gemm<false><<<gg, 256, 0, stream>>>(act0, W[1], Bc[1], act1);
  k_gemm<false><<<gg, 256, 0, stream>>>(act1, W[2], Bc[2], act0);
  k_gemm<true ><<<gg, 256, 0, stream>>>(act0, W[3], bias,  out);
}
